// JetMoeNativeEPWrapper_41987600286209
// MI455X (gfx1250) — matrix operand order repaired and checked by reading, not compiled, not run on hardware
//
#include <hip/hip_runtime.h>
#include <hip/hip_bf16.h>

// ---------------- problem constants ----------------
#define N_TOK 2048
#define DMODEL 2048
#define HFF 5632
#define NEXP 8

typedef __attribute__((ext_vector_type(16))) __bf16 v16bf;
typedef __attribute__((ext_vector_type(8)))  float  v8f;
typedef __attribute__((ext_vector_type(4)))  unsigned int v4u;
typedef __attribute__((ext_vector_type(8)))  int    v8i;
typedef __attribute__((ext_vector_type(4)))  int    v4i;

#if __has_builtin(__builtin_amdgcn_tensor_load_to_lds)
#define USE_TDM 1
#else
#define USE_TDM 0
#endif

// ---------------- workspace layout (bytes) ----------------
#define OFF_XB   ((size_t)0)                               // N*D bf16      =  8388608
#define OFF_ACT  ((size_t)8388608)                         // 2N*H bf16     = 46137344
#define OFF_Y    ((size_t)54525952)                        // 2N*D f32      = 33554432
#define OFF_SG   ((size_t)88080384)                        // 2N f32        =    16384
#define OFF_CNT  ((size_t)88096768)                        // E int (pad)
#define OFF_LT   ((size_t)88097024)                        // E*N int       =    65536
#define OFF_LS   ((size_t)88162560)                        // E*N int       =    65536

// ---------------- helpers ----------------
// 16 consecutive floats -> v16bf via 4x 128-bit loads + packed converts
__device__ __forceinline__ v16bf cvt16(const float* __restrict__ p) {
    union { float4 q[4]; float f[16]; } t;
    t.q[0] = ((const float4*)p)[0];
    t.q[1] = ((const float4*)p)[1];
    t.q[2] = ((const float4*)p)[2];
    t.q[3] = ((const float4*)p)[3];
    v16bf r;
#pragma unroll
    for (int i = 0; i < 16; ++i) r[i] = (__bf16)t.f[i];
    return r;
}

// ---------------- TDM: 2D tile load Global -> LDS ----------------
#if USE_TDM
__device__ __forceinline__ void tdm_load_2d(unsigned lds_off, const void* gptr,
                                            unsigned tensor_d0, unsigned tensor_d1,
                                            unsigned tile_d0, unsigned tile_d1,
                                            unsigned stride_d0, unsigned ds_code) {
    unsigned long long ga = (unsigned long long)(size_t)gptr;
    v4u g0;
    g0[0] = 1u;                                               // count=1, no gather
    g0[1] = lds_off;                                          // lds_addr
    g0[2] = (unsigned)(ga & 0xFFFFFFFFu);                     // global_addr[31:0]
    g0[3] = (unsigned)((ga >> 32) & 0x1FFFFFFu) | (2u << 30); // addr[56:32] | type=2
    v8i g1;
    g1[0] = (int)(ds_code << 16);                             // data_size
    g1[1] = (int)((tensor_d0 & 0xFFFFu) << 16);               // tensor_dim0 lo16
    g1[2] = (int)(((tensor_d0 >> 16) & 0xFFFFu) |
                  ((tensor_d1 & 0xFFFFu) << 16));             // dim0 hi / dim1 lo
    g1[3] = (int)(((tensor_d1 >> 16) & 0xFFFFu) |
                  ((tile_d0 & 0xFFFFu) << 16));               // dim1 hi / tile_dim0
    g1[4] = (int)(tile_d1 & 0xFFFFu);                         // tile_dim1 (tile_dim2=0)
    g1[5] = (int)stride_d0;                                   // tensor_dim0_stride lo32
    g1[6] = 0;
    g1[7] = 0;
    v4i g2 = {0, 0, 0, 0};
    v4i g3 = {0, 0, 0, 0};
#if defined(__clang_major__) && (__clang_major__ >= 23)
    v8i gz = {0, 0, 0, 0, 0, 0, 0, 0};
    __builtin_amdgcn_tensor_load_to_lds(g0, g1, g2, g3, gz, 0);
#else
    __builtin_amdgcn_tensor_load_to_lds(g0, g1, g2, g3, 0);
#endif
}
#endif

// ---------------- tiny init ----------------
__global__ void k_zero_counts(int* counts) {
    if (threadIdx.x < NEXP) counts[threadIdx.x] = 0;
}

// ---------------- fp32 -> bf16 convert of x ----------------
__global__ void k_convert(const float* __restrict__ x, __bf16* __restrict__ xb, int n) {
    int i = blockIdx.x * blockDim.x + threadIdx.x;
    if (i < n) xb[i] = (__bf16)x[i];
}

// ---------------- router: logits, top-2 softmax, dispatch lists ----------------
__global__ void k_router(const float* __restrict__ x, const float* __restrict__ wr,
                         float* __restrict__ logits_out, float* __restrict__ slot_gate,
                         int* __restrict__ counts, int* __restrict__ listTok,
                         int* __restrict__ listSlot) {
    __shared__ float red[256][NEXP];
    int n = blockIdx.x;
    const float* xr = x + (size_t)n * DMODEL;
    float acc[NEXP];
#pragma unroll
    for (int e = 0; e < NEXP; ++e) acc[e] = 0.f;
    for (int d = threadIdx.x; d < DMODEL; d += 256) {
        float xv = xr[d];
#pragma unroll
        for (int e = 0; e < NEXP; ++e) acc[e] += xv * wr[e * DMODEL + d];
    }
#pragma unroll
    for (int e = 0; e < NEXP; ++e) red[threadIdx.x][e] = acc[e];
    __syncthreads();
    for (int s = 128; s > 0; s >>= 1) {
        if (threadIdx.x < (unsigned)s) {
#pragma unroll
            for (int e = 0; e < NEXP; ++e) red[threadIdx.x][e] += red[threadIdx.x + s][e];
        }
        __syncthreads();
    }
    if (threadIdx.x == 0) {
        float lg[NEXP];
#pragma unroll
        for (int e = 0; e < NEXP; ++e) {
            lg[e] = red[0][e];
            logits_out[(size_t)n * NEXP + e] = lg[e];
        }
        int i0 = 0;
#pragma unroll
        for (int e = 1; e < NEXP; ++e) if (lg[e] > lg[i0]) i0 = e;
        int i1 = (i0 == 0) ? 1 : 0;
#pragma unroll
        for (int e = 0; e < NEXP; ++e) if (e != i0 && lg[e] > lg[i1]) i1 = e;
        float e1 = __expf(lg[i1] - lg[i0]);
        float inv = __builtin_amdgcn_rcpf(1.f + e1);
        slot_gate[2 * n + 0] = inv;
        slot_gate[2 * n + 1] = e1 * inv;
        int p0 = atomicAdd(&counts[i0], 1);
        listTok[i0 * N_TOK + p0] = n;  listSlot[i0 * N_TOK + p0] = 2 * n;
        int p1 = atomicAdd(&counts[i1], 1);
        listTok[i1 * N_TOK + p1] = n;  listSlot[i1 * N_TOK + p1] = 2 * n + 1;
    }
}

// ---------------- GEMM1 + SwiGLU (bf16 WMMA, double-buffered TDM B) ----------------
// dyn-LDS: As[2] bf16[16*64] @0/2048 | Bl[2] f32[64*64] @4096/20480
//          | Br[2] f32[64*64] @36864/53248 | tok @69632 | slot @69696
#define G1_SMEM 69760
__global__ __launch_bounds__(128) void k_gemm1(
        const __bf16* __restrict__ xb, const float* __restrict__ win,
        __bf16* __restrict__ act, const int* __restrict__ counts,
        const int* __restrict__ listTok, const int* __restrict__ listSlot) {
    extern __shared__ char smem[];
    int*    tokS  = (int*)(smem + 69632);
    int*    slotS = (int*)(smem + 69696);

    int e   = blockIdx.z;
    int cnt = counts[e];
    int rt  = blockIdx.y;
    if (rt * 16 >= cnt) return;

    int tid = threadIdx.x;
    if (tid < 16) {
        int r = rt * 16 + tid;
        bool v = (r < cnt);
        tokS[tid]  = v ? listTok[e * N_TOK + r]  : listTok[e * N_TOK + rt * 16];
        slotS[tid] = v ? listSlot[e * N_TOK + r] : -1;
    }
    __syncthreads();

    int wave = tid >> 5, lane = tid & 31;
    int nloc = lane & 15;
    int jb   = blockIdx.x * 64;
    int jcol = jb + wave * 16 + nloc;
    const float* wl  = win + (size_t)e * (2 * (size_t)HFF) * DMODEL
                           + (size_t)jb * DMODEL;           // 64-row left tile base
    const float* wrg = wl + (size_t)HFF * DMODEL;           // 64-row right tile base
    int kb0 = (lane < 16) ? 0 : 16;
    int b0  = (lane < 16) ? 0 : 16;
    int b1  = (lane < 16) ? 8 : 24;
    int arow = lane & 15;
    int brow = (wave * 16 + nloc) * 64;

    // preload this lane's 8 output-row slots into registers
    int myslot[8];
#pragma unroll
    for (int i = 0; i < 8; ++i) myslot[i] = slotS[(lane < 16) ? i : (i + 8)];

    // A staging: 128 threads x 8 bf16 (16 bytes) covers 16x64
    int sIdx = tid * 8;
    int sRow = sIdx >> 6, sCol = sIdx & 63;
    const __bf16* xrow = xb + (size_t)tokS[sRow] * DMODEL + sCol;

    const int NIT = DMODEL / 64;
    // prologue: tile 0 into buffer 0
    *(uint4*)(smem + (sRow * 64 + sCol) * 2) = *(const uint4*)(xrow);
#if USE_TDM
    if (wave == 0) {
        tdm_load_2d(4096u,  wl,  DMODEL, 64, 64, 64, DMODEL, 2u);
        tdm_load_2d(36864u, wrg, DMODEL, 64, 64, 64, DMODEL, 2u);
    }
#endif

    v8f accL = {}; v8f accR = {};
    for (int kbi = 0; kbi < NIT; ++kbi) {
        int cur = kbi & 1, nxt = cur ^ 1;
        int kn  = (kbi + 1) * 64;
        const __bf16* Ac  = (const __bf16*)(smem + cur * 2048);
        __bf16*       An  = (__bf16*)(smem + nxt * 2048);
        const float*  Blc = (const float*)(smem + 4096)  + cur * 4096;
        const float*  Brc = (const float*)(smem + 36864) + cur * 4096;
        __syncthreads();                       // prior readers of buf[nxt] done
        if (kbi + 1 < NIT) {
#if USE_TDM
            if (wave == 0) {
                tdm_load_2d(4096u  + (unsigned)nxt * 16384u, wl + kn,
                            DMODEL, 64, 64, 64, DMODEL, 2u);
                tdm_load_2d(36864u + (unsigned)nxt * 16384u, wrg + kn,
                            DMODEL, 64, 64, 64, DMODEL, 2u);
            }
#endif
            *(uint4*)&An[sRow * 64 + sCol] = *(const uint4*)(xrow + kn);
        }
#if USE_TDM
        if (wave == 0) {
            if (kbi + 1 < NIT) __builtin_amdgcn_s_wait_tensorcnt(2); // cur pair done
            else               __builtin_amdgcn_s_wait_tensorcnt(0);
        }
#endif
        __syncthreads();                       // buf[cur] ready for all waves

#pragma unroll
        for (int c = 0; c < 2; ++c) {
            union { uint4 u[2]; v16bf v; } af;
            af.u[0] = *(const uint4*)&Ac[arow * 64 + c * 32 + b0];
            af.u[1] = *(const uint4*)&Ac[arow * 64 + c * 32 + b1];
#if USE_TDM
            v16bf bfl = cvt16(Blc + brow + c * 32 + kb0);
            v16bf bfr = cvt16(Brc + brow + c * 32 + kb0);
#else
            const float* pl = win + (size_t)e * (2 * (size_t)HFF) * DMODEL
                                  + (size_t)jcol * DMODEL + kbi * 64 + c * 32 + kb0;
            v16bf bfl = cvt16(pl);
            v16bf bfr = cvt16(pl + (size_t)HFF * DMODEL);
#endif
            accL = __builtin_amdgcn_wmma_f32_16x16x32_bf16(false, af.v, false, bfl,
                                                           (short)0, accL, false, false);
            accR = __builtin_amdgcn_wmma_f32_16x16x32_bf16(false, af.v, false, bfr,
                                                           (short)0, accR, false, false);
        }
    }

    // SwiGLU + scatter-store (fast sigmoid via v_rcp_f32)
#pragma unroll
    for (int i = 0; i < 8; ++i) {
        int s = myslot[i];
        if (s >= 0) {
            float l = accL[i], r = accR[i];
            float a = l * __builtin_amdgcn_rcpf(1.f + __expf(-l)) * r; // silu(l)*r
            act[(size_t)s * HFF + jcol] = (__bf16)a;
        }
    }
}

// ---------------- GEMM2 (bf16 WMMA, double-buffered TDM B) ----------------
// dyn-LDS: As[2] bf16[16*64] @0/2048 | Bw[2] f32[64*64] @4096/20480
//          | slot @36864 | gslot @36928
#define G2_SMEM 36992
__global__ __launch_bounds__(128) void k_gemm2(
        const __bf16* __restrict__ act, const float* __restrict__ wout,
        float* __restrict__ y, const int* __restrict__ counts,
        const int* __restrict__ listSlot) {
    extern __shared__ char smem[];
    int*    slotS  = (int*)(smem + 36864);
    int*    gslotS = (int*)(smem + 36928);

    int e   = blockIdx.z;
    int cnt = counts[e];
    int rt  = blockIdx.y;
    if (rt * 16 >= cnt) return;

    int tid = threadIdx.x;
    if (tid < 16) {
        int r = rt * 16 + tid;
        bool v = (r < cnt);
        slotS[tid]  = v ? listSlot[e * N_TOK + r] : -1;
        gslotS[tid] = v ? listSlot[e * N_TOK + r] : listSlot[e * N_TOK + rt * 16];
    }
    __syncthreads();

    int wave = tid >> 5, lane = tid & 31;
    int nloc = lane & 15;
    int db   = blockIdx.x * 64;
    int dcol = db + wave * 16 + nloc;
    const float* wb = wout + (size_t)e * DMODEL * (size_t)HFF
                           + (size_t)db * HFF;              // 64-row tile base
    int kb0 = (lane < 16) ? 0 : 16;
    int b0  = (lane < 16) ? 0 : 16;
    int b1  = (lane < 16) ? 8 : 24;
    int arow = lane & 15;
    int brow = (wave * 16 + nloc) * 64;

    int myslot[8];
#pragma unroll
    for (int i = 0; i < 8; ++i) myslot[i] = slotS[(lane < 16) ? i : (i + 8)];

    int sIdx = tid * 8;
    int sRow = sIdx >> 6, sCol = sIdx & 63;
    const __bf16* arowp = act + (size_t)gslotS[sRow] * HFF + sCol;

    const int NIT = HFF / 64;
    *(uint4*)(smem + (sRow * 64 + sCol) * 2) = *(const uint4*)(arowp);
#if USE_TDM
    if (wave == 0) tdm_load_2d(4096u, wb, HFF, 64, 64, 64, HFF, 2u);
#endif

    v8f acc = {};
    for (int kbi = 0; kbi < NIT; ++kbi) {
        int cur = kbi & 1, nxt = cur ^ 1;
        int kn  = (kbi + 1) * 64;
        const __bf16* Ac  = (const __bf16*)(smem + cur * 2048);
        __bf16*       An  = (__bf16*)(smem + nxt * 2048);
        const float*  Bwc = (const float*)(smem + 4096) + cur * 4096;
        __syncthreads();
        if (kbi + 1 < NIT) {
#if USE_TDM
            if (wave == 0)
                tdm_load_2d(4096u + (unsigned)nxt * 16384u, wb + kn,
                            HFF, 64, 64, 64, HFF, 2u);
#endif
            *(uint4*)&An[sRow * 64 + sCol] = *(const uint4*)(arowp + kn);
        }
#if USE_TDM
        if (wave == 0) {
            if (kbi + 1 < NIT) __builtin_amdgcn_s_wait_tensorcnt(1);
            else               __builtin_amdgcn_s_wait_tensorcnt(0);
        }
#endif
        __syncthreads();

#pragma unroll
        for (int c = 0; c < 2; ++c) {
            union { uint4 u[2]; v16bf v; } af;
            af.u[0] = *(const uint4*)&Ac[arow * 64 + c * 32 + b0];
            af.u[1] = *(const uint4*)&Ac[arow * 64 + c * 32 + b1];
#if USE_TDM
            v16bf bfb = cvt16(Bwc + brow + c * 32 + kb0);
#else
            v16bf bfb = cvt16(wout + (size_t)e * DMODEL * (size_t)HFF
                                   + (size_t)dcol * HFF + kbi * 64 + c * 32 + kb0);
#endif
            acc = __builtin_amdgcn_wmma_f32_16x16x32_bf16(false, af.v, false, bfb,
                                                          (short)0, acc, false, false);
        }
    }

#pragma unroll
    for (int i = 0; i < 8; ++i) {
        int s = myslot[i];
        if (s >= 0) y[(size_t)s * DMODEL + dcol] = acc[i];
    }
}

// ---------------- combine: out = bias + g0*y[2n] + g1*y[2n+1] ----------------
__global__ void k_combine(const float* __restrict__ y, const float* __restrict__ sg,
                          const float* __restrict__ bias, float* __restrict__ out) {
    size_t i = (size_t)blockIdx.x * blockDim.x + threadIdx.x;
    if (i >= (size_t)N_TOK * DMODEL) return;
    int n = (int)(i / DMODEL);
    int d = (int)(i % DMODEL);
    float g0 = sg[2 * n + 0], g1 = sg[2 * n + 1];
    out[i] = bias[d] + g0 * y[(size_t)(2 * n) * DMODEL + d]
                     + g1 * y[(size_t)(2 * n + 1) * DMODEL + d];
}

// ---------------- launch ----------------
extern "C" void kernel_launch(void* const* d_in, const int* in_sizes, int n_in,
                              void* d_out, int out_size, void* d_ws, size_t ws_size,
                              hipStream_t stream) {
    const float* x    = (const float*)d_in[0];   // [N, D]
    const float* wr   = (const float*)d_in[1];   // [E, D]
    const float* win  = (const float*)d_in[2];   // [E, 2H, D]
    const float* wout = (const float*)d_in[3];   // [E, D, H]
    const float* bias = (const float*)d_in[4];   // [D]

    float* out    = (float*)d_out;                       // [N*D] layer_output
    float* logits = out + (size_t)N_TOK * DMODEL;        // [N*E] router_logits

    char* ws = (char*)d_ws;
    __bf16* xb      = (__bf16*)(ws + OFF_XB);
    __bf16* act     = (__bf16*)(ws + OFF_ACT);
    float*  y       = (float*) (ws + OFF_Y);
    float*  sg      = (float*) (ws + OFF_SG);
    int*    counts  = (int*)   (ws + OFF_CNT);
    int*    listTok = (int*)   (ws + OFF_LT);
    int*    listSlot= (int*)   (ws + OFF_LS);

    k_zero_counts<<<1, 32, 0, stream>>>(counts);

    int nelem = N_TOK * DMODEL;
    k_convert<<<(nelem + 255) / 256, 256, 0, stream>>>(x, xb, nelem);

    k_router<<<N_TOK, 256, 0, stream>>>(x, wr, logits, sg, counts, listTok, listSlot);

    dim3 g1(HFF / 64, N_TOK / 16, NEXP);
    k_gemm1<<<g1, 128, G1_SMEM, stream>>>(xb, win, act, counts, listTok, listSlot);

    dim3 g2(DMODEL / 64, N_TOK / 16, NEXP);
    k_gemm2<<<g2, 128, G2_SMEM, stream>>>(act, wout, y, counts, listSlot);

    k_combine<<<(nelem + 255) / 256, 256, 0, stream>>>(y, sg, bias, out);
}
